// PairwiseLengthAttn1D_71279277244737
// MI455X (gfx1250) — matrix operand order repaired and checked by reading, not compiled, not run on hardware
//
#include <hip/hip_runtime.h>
#include <hip/hip_bf16.h>
#include <math.h>

typedef __attribute__((ext_vector_type(16))) __bf16 v16bf;
typedef __attribute__((ext_vector_type(8)))  float  v8f;

union Frag { v16bf v; unsigned u[8]; };

static __device__ __forceinline__ unsigned short f2bfbits(float f) {
    unsigned u = __builtin_bit_cast(unsigned, f);
    unsigned r = (u + 0x7fffu + ((u >> 16) & 1u)) >> 16;   // RNE
    return (unsigned short)r;
}
static __device__ __forceinline__ __bf16 f2bf(float f) {
    unsigned short s = f2bfbits(f);
    return __builtin_bit_cast(__bf16, s);
}
static __device__ __forceinline__ unsigned pack_bf2(float lo, float hi) {
    return ((unsigned)f2bfbits(hi) << 16) | (unsigned)f2bfbits(lo);
}
static __device__ __forceinline__ float gelu_exact(float x) {
    return 0.5f * x * (1.0f + erff(x * 0.70710678118654752f));
}

// ---------------- constants ----------------
#define NB   8
#define CC   256
#define TT   512
#define HH   8
#define NH   64          // N*H
#define DK   32
#define DV   32
#define TWOT 1024
#define MM   32768       // NH * TT

// ---- CDNA5 async global->LDS copy (ASYNCcnt path), with portable fallback ----
#if __has_builtin(__builtin_amdgcn_global_load_async_to_lds_b128) && \
    __has_builtin(__builtin_amdgcn_s_wait_asynccnt)
#define USE_ASYNC_LDS 1
#else
#define USE_ASYNC_LDS 0
#endif

typedef int v4i_vec __attribute__((vector_size(16)));
typedef __attribute__((address_space(1))) v4i_vec* gv4i_p;   // global int4*
typedef __attribute__((address_space(3))) v4i_vec* lv4i_p;   // LDS int4*

#if USE_ASYNC_LDS
static __device__ __forceinline__ void async_copy16(const __bf16* g, __bf16* l) {
    __builtin_amdgcn_global_load_async_to_lds_b128((gv4i_p)g, (lv4i_p)l, 0, 0);
}
#endif
static __device__ __forceinline__ void sync_copy32(const __bf16* g, __bf16* l) {
    const uint4* gp = (const uint4*)g;
    uint4 a = gp[0], b = gp[1];
    uint4* lp = (uint4*)l;
    lp[0] = a; lp[1] = b;
}

// ---------------- circulant weight expansion, TRANSPOSED: cpT[o][k] bf16 -----
// cpT[o][2t+r] = w[r][(o - t*stride) mod od]  ;  cpT is (od, 1024)
__global__ void build_cpT(const float* __restrict__ w, __bf16* __restrict__ cpT,
                          int od, int stride) {
    int g = blockIdx.x * 256 + threadIdx.x;
    int total = TWOT * od;
    if (g >= total) return;
    int row = g & (TWOT - 1);       // k index (2t+r), coalesced over threads
    int o   = g >> 10;
    int t = row >> 1, r = row & 1;
    int src = o - t * stride;
    src %= od; if (src < 0) src += od;
    cpT[(size_t)o * TWOT + row] = f2bf(w[r * od + src]);
}

// ---------------- qkv projection: out = W(256x256) @ x[n] ---------------------
// mode 0/2: out[(nh*T + t)*32 + d]   (q is pre-scaled by 1/sqrt(dk))
// mode 1:   out[(nh*32 + d)*T + t]
__global__ __launch_bounds__(256) void qkv_proj(const float* __restrict__ x,
                                                const float* __restrict__ W,
                                                float* __restrict__ out,
                                                int mode, float scale) {
    int g = blockIdx.x * 256 + threadIdx.x;      // N*256*512 = 1M threads
    int t = g & (TT - 1);
    int o = (g >> 9) & 255;
    int n = g >> 17;
    float acc = 0.f;
    const float* xr = x + (size_t)n * CC * TT + t;
    const float* wr = W + (size_t)o * CC;
    #pragma unroll 4
    for (int c = 0; c < CC; ++c) acc += wr[c] * xr[(size_t)c * TT];
    acc *= scale;
    int h = o >> 5, d = o & 31;
    int nh = n * HH + h;
    if (mode == 1) out[((size_t)nh * 32 + d) * TT + t] = acc;
    else           out[((size_t)nh * TT + t) * 32 + d] = acc;
}

// ---------------- scores + positional channel pack (bf16 output) -------------
// p[nh][t][2s] = (q.k)/sqrt(dk) ; p[nh][t][2s+1] = |t-s|/(T-1)  (packed pairs)
__global__ __launch_bounds__(256) void scores_pack(const float* __restrict__ q,
                                                   const float* __restrict__ k2,
                                                   unsigned* __restrict__ p) {
    int t = blockIdx.x, nh = blockIdx.y;
    __shared__ float qrow[DK];
    if (threadIdx.x < DK) qrow[threadIdx.x] = q[((size_t)nh * TT + t) * DK + threadIdx.x];
    __syncthreads();
    for (int s = threadIdx.x; s < TT; s += 256) {
        float acc = 0.f;
        #pragma unroll 8
        for (int d = 0; d < DK; ++d)
            acc += qrow[d] * k2[((size_t)nh * DK + d) * TT + s];
        float pd = fabsf((float)(t - s)) * (1.0f / (float)(TT - 1));
        p[((size_t)nh * TT + t) * (TWOT / 2) + s] = pack_bf2(acc, pd);
    }
}

// ---------------- bf16 WMMA GEMM: C(M,Nn) = A(M,K) * Bt(Nn,K)^T --------------
// A bf16 row-major (M,K); Bt bf16 (Nn,K) [pre-transposed weights]; C fp32.
// 128x128 block tile, 8 waves (4x2), each wave 32x64 -> 8 wmma accumulators.
#define TM 128
#define TN 128
#define LDSW 48   // LDS row stride in bf16 elements (96B: keeps 16B chunks aligned)
__global__ __launch_bounds__(256) void gemm_bf16(const __bf16* __restrict__ A,
                                                 const __bf16* __restrict__ Bt,
                                                 float* __restrict__ C,
                                                 int K, int Nn,
                                                 float* __restrict__ part,
                                                 int do_stats) {
    __shared__ __bf16 sA[TM * LDSW];   // [m][k]
    __shared__ __bf16 sB[TN * LDSW];   // [n][k]
    __shared__ float red1[256], red2[256];

    int tid  = threadIdx.x;
    int lane = tid & 31, wid = tid >> 5;
    int wm = wid & 3, wn = wid >> 2;            // 4 x 2 wave grid
    int gm0 = blockIdx.x * TM, gn0 = blockIdx.y * TN;

    v8f zero = {0.f,0.f,0.f,0.f,0.f,0.f,0.f,0.f};
    v8f acc[2][4];
    #pragma unroll
    for (int i = 0; i < 2; ++i)
        #pragma unroll
        for (int j = 0; j < 4; ++j) acc[i][j] = zero;

    int srow = tid >> 1, sseg = (tid & 1) * 16;    // 128 rows x 2 segs of 16 elems

    for (int kk = 0; kk < K; kk += 32) {
        __syncthreads();                           // LDS reusable (all waves done)
        const __bf16* ga = A  + (size_t)(gm0 + srow) * K + kk + sseg;
        const __bf16* gb = Bt + (size_t)(gn0 + srow) * K + kk + sseg;
        __bf16* la = &sA[srow * LDSW + sseg];
        __bf16* lb = &sB[srow * LDSW + sseg];
#if USE_ASYNC_LDS
        async_copy16(ga,     la);
        async_copy16(ga + 8, la + 8);
        async_copy16(gb,     lb);
        async_copy16(gb + 8, lb + 8);
#else
        sync_copy32(ga, la);
        sync_copy32(gb, lb);
#endif
        if (kk + 32 < K) {
            __builtin_prefetch(ga + 32, 0, 1);
            __builtin_prefetch(gb + 32, 0, 1);
        }
#if USE_ASYNC_LDS
        __builtin_amdgcn_s_wait_asynccnt(0);
#endif
        __syncthreads();

        const unsigned* sa = (const unsigned*)sA;
        const unsigned* sb = (const unsigned*)sB;
        int half = lane >> 4;                 // 0/1 per ISA A/B fragment layout
        Frag fa[2], fb[4];
        #pragma unroll
        for (int i = 0; i < 2; ++i) {
            int m = wm * 32 + i * 16 + (lane & 15);
            #pragma unroll
            for (int v = 0; v < 8; ++v) {
                int kid = (v < 4 ? 0 : 8) + half * 4 + (v & 3);   // packed k-pair idx
                fa[i].u[v] = sa[m * (LDSW / 2) + kid];
            }
        }
        #pragma unroll
        for (int j = 0; j < 4; ++j) {
            int n = wn * 64 + j * 16 + (lane & 15);
            #pragma unroll
            for (int v = 0; v < 8; ++v)
                fb[j].u[v] = sb[n * (LDSW / 2) + (v < 4 ? 0 : 8) + half * 4 + (v & 3)];
        }
        #pragma unroll
        for (int i = 0; i < 2; ++i)
            #pragma unroll
            for (int j = 0; j < 4; ++j)
                acc[i][j] = __builtin_amdgcn_wmma_f32_16x16x32_bf16(
                    false, fa[i].v, false, fb[j].v, (short)0, acc[i][j], false, false);
    }

    // epilogue: store + (optional) per-block LN partial sums
    float lsum = 0.f, lsq = 0.f;
    #pragma unroll
    for (int i = 0; i < 2; ++i)
        #pragma unroll
        for (int j = 0; j < 4; ++j)
            #pragma unroll
            for (int r = 0; r < 8; ++r) {
                float val = acc[i][j][r];
                int row = gm0 + wm * 32 + i * 16 + r + ((lane >> 4) << 3);
                int col = gn0 + wn * 64 + j * 16 + (lane & 15);
                C[(size_t)row * Nn + col] = val;
                lsum += val; lsq += val * val;
            }
    if (do_stats) {
        red1[tid] = lsum; red2[tid] = lsq;
        __syncthreads();
        for (int s = 128; s > 0; s >>= 1) {
            if (tid < s) { red1[tid] += red1[tid + s]; red2[tid] += red2[tid + s]; }
            __syncthreads();
        }
        if (tid == 0) {
            int bid = blockIdx.x * gridDim.y + blockIdx.y;
            part[2 * bid]     = red1[0];
            part[2 * bid + 1] = red2[0];
        }
    }
}

// ---------------- deterministic LN stats reduce ------------------------------
__global__ void stats_reduce(const float* __restrict__ part, float* __restrict__ stats, int Ny) {
    int nh = threadIdx.x;
    if (nh >= NH) return;
    float s = 0.f, q = 0.f;
    for (int mb = 0; mb < 4; ++mb)
        for (int by = 0; by < Ny; ++by) {
            int bid = (nh * 4 + mb) * Ny + by;
            s += part[2 * bid];
            q += part[2 * bid + 1];
        }
    stats[2 * nh] = s;
    stats[2 * nh + 1] = q;
}

// ---------------- LayerNorm + affine + GELU + pair-transpose -----------------
// out(bf16 pairs)[nh][a][2b+r] = gelu( (in[nh][b][2a+r]-mu)*rstd*g[b][2a+r] + bias[b][2a+r] )
__global__ __launch_bounds__(256) void ln_gelu_t(const float* __restrict__ in,
                                                 unsigned* __restrict__ out,
                                                 const float* __restrict__ stats,
                                                 const float* __restrict__ g,
                                                 const float* __restrict__ b) {
    int nh = blockIdx.z;
    int a0 = blockIdx.x * 32, b0 = blockIdx.y * 32;
    __shared__ float2 tile[32][33];
    const float inv_cnt = 1.0f / (float)(TT * TWOT);
    float mean = stats[2 * nh] * inv_cnt;
    float var  = stats[2 * nh + 1] * inv_cnt - mean * mean;
    float rstd = rsqrtf(var + 1e-5f);
    const float2* in2 = (const float2*)in;
    const float2* g2  = (const float2*)g;
    const float2* b2  = (const float2*)b;
    #pragma unroll
    for (int it = 0; it < 4; ++it) {
        int idx = threadIdx.x + it * 256;
        int bb = idx >> 5, aa = idx & 31;
        int row = b0 + bb, col = a0 + aa;
        float2 val = in2[((size_t)nh * TT + row) * (TWOT / 2) + col];
        float2 gg  = g2[(size_t)row * (TWOT / 2) + col];
        float2 bs  = b2[(size_t)row * (TWOT / 2) + col];
        float x0 = (val.x - mean) * rstd * gg.x + bs.x;
        float x1 = (val.y - mean) * rstd * gg.y + bs.y;
        float2 r; r.x = gelu_exact(x0); r.y = gelu_exact(x1);
        tile[bb][aa] = r;
    }
    __syncthreads();
    #pragma unroll
    for (int it = 0; it < 4; ++it) {
        int idx = threadIdx.x + it * 256;
        int aa = idx >> 5, bb = idx & 31;
        float2 r = tile[bb][aa];
        out[((size_t)nh * TT + a0 + aa) * (TWOT / 2) + (b0 + bb)] = pack_bf2(r.x, r.y);
    }
}

// ---------------- final column softmax + attn @ v ----------------------------
// out[n, h*32+d, i] = sum_t softmax_t(pre[nh][t][i]) * v[nh][t][d]
__global__ __launch_bounds__(256) void softmax_av(const float* __restrict__ pre,
                                                  const float* __restrict__ vbuf,
                                                  float* __restrict__ out) {
    int nh = blockIdx.y;
    int i0 = blockIdx.x * 16;
    __shared__ float w[TT * 17];
    __shared__ float red[256];
    __shared__ float col_max[16], col_rsum[16];

    for (int p = 0; p < 32; ++p) {
        int idx = threadIdx.x + p * 256;          // 8192 elements
        int ii = idx & 15, t = idx >> 4;
        w[t * 17 + ii] = pre[((size_t)nh * TT + t) * TT + i0 + ii];
    }
    __syncthreads();

    int ii = threadIdx.x & 15, sub = threadIdx.x >> 4;
    float mx = -1e30f;
    for (int t = sub; t < TT; t += 16) mx = fmaxf(mx, w[t * 17 + ii]);
    red[threadIdx.x] = mx;
    __syncthreads();
    if (sub == 0) {
        float m = red[ii];
        #pragma unroll
        for (int j = 1; j < 16; ++j) m = fmaxf(m, red[j * 16 + ii]);
        col_max[ii] = m;
    }
    __syncthreads();
    float m = col_max[ii];
    float s = 0.f;
    for (int t = sub; t < TT; t += 16) {
        float e = __expf(w[t * 17 + ii] - m);
        w[t * 17 + ii] = e;
        s += e;
    }
    red[threadIdx.x] = s;
    __syncthreads();
    if (sub == 0) {
        float ss = red[ii];
        #pragma unroll
        for (int j = 1; j < 16; ++j) ss += red[j * 16 + ii];
        col_rsum[ii] = 1.0f / ss;
    }
    __syncthreads();

    int n = nh >> 3, h = nh & 7;
    #pragma unroll
    for (int p = 0; p < 2; ++p) {
        int o = threadIdx.x + p * 256;            // 512 outputs: (i,d)
        int d = o & 31, i = o >> 5;
        float acc = 0.f;
        for (int t = 0; t < TT; ++t)
            acc += w[t * 17 + i] * vbuf[((size_t)nh * TT + t) * DV + d];
        acc *= col_rsum[i];
        out[((size_t)n * 256 + h * 32 + d) * TT + (i0 + i)] = acc;
    }
}

// ============================ host launch ====================================
extern "C" void kernel_launch(void* const* d_in, const int* in_sizes, int n_in,
                              void* d_out, int out_size, void* d_ws, size_t ws_size,
                              hipStream_t stream) {
    const float* x    = (const float*)d_in[0];
    const float* Wq   = (const float*)d_in[1];
    const float* Wk   = (const float*)d_in[2];
    const float* Wv   = (const float*)d_in[3];
    const float* m1w0 = (const float*)d_in[4];
    const float* m2w0 = (const float*)d_in[5];
    const float* m1w1 = (const float*)d_in[6];
    const float* m2w1 = (const float*)d_in[7];
    const float* ln1g0 = (const float*)d_in[8];
    const float* ln1b0 = (const float*)d_in[9];
    const float* ln2g0 = (const float*)d_in[10];
    const float* ln2b0 = (const float*)d_in[11];
    const float* ln1g1 = (const float*)d_in[12];
    const float* ln1b1 = (const float*)d_in[13];
    float* outp = (float*)d_out;

    char* ws = (char*)d_ws;
    size_t off = 0;
    __bf16* p0   = (__bf16*)(ws + off); off += (size_t)MM * TWOT * sizeof(__bf16); // 64 MB (activations)
    float*  p1   = (float*)(ws + off);  off += (size_t)MM * TWOT * sizeof(float);  // 128 MB (GEMM out)
    float*  qbuf = (float*)(ws + off);  off += (size_t)NH * TT * DK * sizeof(float);
    float*  kbuf = (float*)(ws + off);  off += (size_t)NH * DK * TT * sizeof(float);
    float*  vbuf = (float*)(ws + off);  off += (size_t)NH * TT * DV * sizeof(float);
    __bf16* cp0  = (__bf16*)(ws + off); off += (size_t)TWOT * TWOT * sizeof(__bf16);
    __bf16* cp1  = (__bf16*)(ws + off); off += (size_t)TWOT * TWOT * sizeof(__bf16);
    __bf16* cp2  = (__bf16*)(ws + off); off += (size_t)TWOT * TWOT * sizeof(__bf16);
    __bf16* cp3  = (__bf16*)(ws + off); off += (size_t)TT * TWOT * sizeof(__bf16);
    float*  part = (float*)(ws + off);  off += (size_t)2048 * 2 * sizeof(float);
    float*  stats= (float*)(ws + off);  off += (size_t)NH * 2 * sizeof(float);
    (void)ws_size; (void)in_sizes; (void)n_in; (void)out_size;

    // 1) circulant weight expansion -> transposed bf16 (Nn, K) matrices
    build_cpT<<<(TWOT * TWOT + 255) / 256, 256, 0, stream>>>(m1w0, cp0, TWOT, 2);
    build_cpT<<<(TWOT * TWOT + 255) / 256, 256, 0, stream>>>(m2w0, cp1, TWOT, 2);
    build_cpT<<<(TWOT * TWOT + 255) / 256, 256, 0, stream>>>(m1w1, cp2, TWOT, 2);
    build_cpT<<<(TWOT * TT   + 255) / 256, 256, 0, stream>>>(m2w1, cp3, TT,   1);

    // 2) q,k,v projections (q pre-scaled by 1/sqrt(dk))
    const int qkv_blocks = (NB * 256 * TT) / 256;
    qkv_proj<<<qkv_blocks, 256, 0, stream>>>(x, Wq, qbuf, 0, 0.17677669529663687f);
    qkv_proj<<<qkv_blocks, 256, 0, stream>>>(x, Wk, kbuf, 1, 1.0f);
    qkv_proj<<<qkv_blocks, 256, 0, stream>>>(x, Wv, vbuf, 2, 1.0f);

    // 3) scores + positional channel -> p0 (bf16 pairs)
    scores_pack<<<dim3(TT, NH), 256, 0, stream>>>(qbuf, kbuf, (unsigned*)p0);

    dim3 gemm_grid(MM / TM, TWOT / TN);       // 256 x 8
    dim3 lng_grid(16, 16, NH);

    // 4) mix block 0
    gemm_bf16<<<gemm_grid, 256, 0, stream>>>(p0, cp0, p1, TWOT, TWOT, part, 1);
    stats_reduce<<<1, 64, 0, stream>>>(part, stats, TWOT / TN);
    ln_gelu_t<<<lng_grid, 256, 0, stream>>>(p1, (unsigned*)p0, stats, ln1g0, ln1b0);

    gemm_bf16<<<gemm_grid, 256, 0, stream>>>(p0, cp1, p1, TWOT, TWOT, part, 1);
    stats_reduce<<<1, 64, 0, stream>>>(part, stats, TWOT / TN);
    ln_gelu_t<<<lng_grid, 256, 0, stream>>>(p1, (unsigned*)p0, stats, ln2g0, ln2b0);

    // 5) mix block 1
    gemm_bf16<<<gemm_grid, 256, 0, stream>>>(p0, cp2, p1, TWOT, TWOT, part, 1);
    stats_reduce<<<1, 64, 0, stream>>>(part, stats, TWOT / TN);
    ln_gelu_t<<<lng_grid, 256, 0, stream>>>(p1, (unsigned*)p0, stats, ln1g1, ln1b1);

    // final sym_linear (out_dim = T, stride 1), no LN
    gemm_bf16<<<dim3(MM / TM, TT / TN), 256, 0, stream>>>(p0, cp3, p1, TWOT, TT, (float*)nullptr, 0);

    // 6) column softmax (== transpose + row softmax) fused with attn @ v
    softmax_av<<<dim3(TT / 16, NH), 256, 0, stream>>>(p1, vbuf, outp);
}
